// SpMiddleAlinPreFramesNoDownsampleXY_21139829031372
// MI455X (gfx1250) — matrix operand order repaired and checked by reading, not compiled, not run on hardware
//
#include <hip/hip_runtime.h>
#include <hip/hip_bf16.h>

// ---------------------------------------------------------------------------
// SpMiddle 3D sparse-conv stack for MI455X (gfx1250).
// Implicit-GEMM conv via v_wmma_f32_16x16x32_bf16, per-tap weight tiles
// double-buffered in LDS via GLOBAL_LOAD_ASYNC_TO_LDS_B128 (ASYNCcnt).
// ---------------------------------------------------------------------------

typedef unsigned short ushort_t;
typedef __attribute__((ext_vector_type(16))) __bf16 v16bf;
typedef __attribute__((ext_vector_type(8)))  float  v8f;
typedef __attribute__((ext_vector_type(4)))  int    v4i;

#if __has_builtin(__builtin_amdgcn_global_load_async_to_lds_b128) && \
    __has_builtin(__builtin_amdgcn_s_wait_asynccnt)
#define USE_ASYNC_LDS 1
typedef __attribute__((address_space(1))) v4i glb_v4i;   // global v4i
typedef __attribute__((address_space(3))) v4i lds_v4i;   // LDS v4i
#else
#define USE_ASYNC_LDS 0
#endif

#define HH 96
#define WW 96
#define BB 2

// fp32 -> bf16 round-to-nearest-even
__device__ __forceinline__ ushort_t f2bf(float f) {
  unsigned int u = __float_as_uint(f);
  unsigned int r = (u + 0x7fffu + ((u >> 16) & 1u)) >> 16;
  return (ushort_t)r;
}

struct __align__(16) B128 { unsigned int w[4]; };

// Two 8-bf16 halves: elements 0..7 from p0, 8..15 from p1 (not used here)
__device__ __forceinline__ v16bf ld_frag2x8(const ushort_t* p0, const ushort_t* p1) {
  v16bf v;
  B128* d = (B128*)&v;
  d[0] = *(const B128*)p0;
  d[1] = *(const B128*)p1;
  return v;
}
// B fragment: lane holds 16 contiguous bf16 (K0+khalf*16..+16)
__device__ __forceinline__ v16bf ld_frag16(const ushort_t* p) {
  v16bf v;
  B128* d = (B128*)&v;
  d[0] = *(const B128*)(p);
  d[1] = *(const B128*)(p + 8);
  return v;
}

// ---------------------------------------------------------------------------
// One-shot per layer: W [taps][CI][CO] f32 -> Wbf [taps][CO][CIP] bf16
// (transposed so B fragments are K-contiguous; K zero-padded to CIP).
// ---------------------------------------------------------------------------
__global__ void prep_weights(const float* __restrict__ W, ushort_t* __restrict__ Wbf,
                             int CI, int CO, int CIP, int NTAPS) {
  int i = blockIdx.x * blockDim.x + threadIdx.x;
  const int total = NTAPS * CO * CIP;
  if (i >= total) return;
  int k = i % CIP; int r = i / CIP;
  int co = r % CO; int t = r / CO;
  float v = (k < CI) ? W[((size_t)t * CI + k) * CO + co] : 0.0f;
  Wbf[i] = f2bf(v);
}

// ---------------------------------------------------------------------------
// Implicit-GEMM conv3d: X [B,DIN,96,96,CI] * Wbf [taps][CO][CIP] -> Y [B,DOUT,96,96,CO]
// Block: 256 threads (8 waves), computes 8x8 spatial x CO for one (b, z_out).
// ---------------------------------------------------------------------------
template<int CI, int CO, int SD, int PDZ, int KD, int KH, int KW>
__global__ __launch_bounds__(256)
void conv3d_wmma(const float* __restrict__ X, const ushort_t* __restrict__ Wbf,
                 float* __restrict__ Y, int DIN, int DOUT)
{
  constexpr int CIP    = (CI < 32) ? 32 : CI;      // K padded to one WMMA step
  constexpr int PADC   = (CIP > CI) ? (CIP - CI) : 1; // safe divisor
  constexpr int PY     = 8 + KH - 1;
  constexpr int PX     = 8 + KW - 1;
  constexpr int OY     = (KH > 1) ? 1 : 0;         // y/x pad (1 for 3-tap, 0 for 1-tap)
  constexpr int OX     = (KW > 1) ? 1 : 0;
  constexpr int NT     = CO / 16;                  // N tiles (2 or 4)
  constexpr int NTW    = (NT >= 2) ? (NT / 2) : 1; // N tiles per wave
  constexpr int KSTEPS = CIP / 32;
  constexpr int NTAPS  = KD * KH * KW;
  constexpr int WTILE  = CO * CIP;                 // bf16 elements per tap tile
  constexpr int WCHUNK = WTILE / 8;                // 16-byte chunks per tap tile

  __shared__ __align__(16) ushort_t sA[KD * PY * PX * CIP]; // input patch (bf16)
  __shared__ __align__(16) ushort_t sB[2][WTILE];           // double-buffered weights

  int blk = blockIdx.x;
  const int tx = blk % (WW / 8); blk /= (WW / 8);
  const int ty = blk % (HH / 8); blk /= (HH / 8);
  const int zo = blk % DOUT;
  const int b  = blk / DOUT;
  const int y0 = ty * 8, x0 = tx * 8;
  const int zi0 = zo * SD - PDZ;
  const int tid = threadIdx.x;

  // Zero the K-pad region of sA once (CI=16 only).
  if constexpr (CIP > CI) {
    for (int e = tid; e < KD * PY * PX * PADC; e += 256) {
      int r = e / PADC; int k = CI + e % PADC;
      sA[r * CIP + k] = 0;
    }
  }

  // Stage input patch f32 -> bf16 (float4 granularity, zero outside grid).
  {
    constexpr int C4 = CI / 4;
    for (int e = tid; e < KD * PY * PX * C4; e += 256) {
      int c4 = e % C4; int r = e / C4;
      int xx = r % PX; r /= PX;
      int yy = r % PY; int kd = r / PY;
      int iz = zi0 + kd;
      int iy = y0 - OY + yy;
      int ix = x0 - OX + xx;
      ushort_t* dst = &sA[((kd * PY + yy) * PX + xx) * CIP + c4 * 4];
      if (iz >= 0 && iz < DIN && iy >= 0 && iy < HH && ix >= 0 && ix < WW) {
        const float4 v = *(const float4*)&X[(size_t)((((b * DIN + iz) * HH) + iy) * WW + ix) * CI + c4 * 4];
        dst[0] = f2bf(v.x); dst[1] = f2bf(v.y); dst[2] = f2bf(v.z); dst[3] = f2bf(v.w);
      } else {
        dst[0] = 0; dst[1] = 0; dst[2] = 0; dst[3] = 0;
      }
    }
  }

  // Per-tap weight tile staging: contiguous b128 chunks, async into LDS.
  auto stage_tap = [&](int t, int bufsel) {
    const ushort_t* src = Wbf + (size_t)t * WTILE;
#if USE_ASYNC_LDS
    for (int c2 = tid; c2 < WCHUNK; c2 += 256) {
      __builtin_amdgcn_global_load_async_to_lds_b128(
          (glb_v4i*)(src + c2 * 8), (lds_v4i*)(&sB[bufsel][c2 * 8]), 0, 0);
    }
#else
    for (int c2 = tid; c2 < WCHUNK; c2 += 256) {
      *(B128*)&sB[bufsel][c2 * 8] = *(const B128*)(src + c2 * 8);
    }
#endif
  };

  // Wave / lane -> tile mapping (per ISA VGPR layouts).
  const int wv    = tid >> 5;
  const int lane  = tid & 31;
  const int mrow  = lane & 15;
  const int khalf = lane >> 4;
  const int mtile = wv & 3;
  const int nt0   = (wv >> 2) * NTW;
  const int p  = mtile * 16 + mrow;   // local output position 0..63
  const int py = p >> 3, px = p & 7;

  v8f acc[NTW];
#pragma unroll
  for (int j = 0; j < NTW; ++j)
#pragma unroll
    for (int k = 0; k < 8; ++k) acc[j][k] = 0.0f;

  stage_tap(0, 0);
  for (int t = 0; t < NTAPS; ++t) {
#if USE_ASYNC_LDS
    __builtin_amdgcn_s_wait_asynccnt(0);   // tap t's tile landed (own wave)
#endif
    __syncthreads();                        // all waves: tile t (and sA) visible
    if (t + 1 < NTAPS) stage_tap(t + 1, (t + 1) & 1); // overlap with compute

    const ushort_t* sBt = sB[t & 1];
    const int kd = t / (KH * KW), kh = (t / KW) % KH, kw = t % KW;
    const ushort_t* arow = &sA[(((kd * PY) + py + kh) * PX + px + kw) * CIP];
#pragma unroll
    for (int ks = 0; ks < KSTEPS; ++ks) {
      const int K0 = ks * 32;
      v16bf a = ld_frag16(arow + K0 + khalf * 16);
#pragma unroll
      for (int j = 0; j < NTW; ++j) {
        const int n = (nt0 + j) * 16 + mrow;
        v16bf bf = ld_frag16(&sBt[n * CIP + K0 + khalf * 16]);
        acc[j] = __builtin_amdgcn_wmma_f32_16x16x32_bf16(
            /*neg_a=*/false, a, /*neg_b=*/false, bf,
            /*c_mod=*/(short)0, acc[j], /*reuse_a=*/false, /*reuse_b=*/false);
      }
    }
  }

  // D layout: VGPR v -> row khalf*8+v, col lane&15.
#pragma unroll
  for (int j = 0; j < NTW; ++j) {
#pragma unroll
    for (int v = 0; v < 8; ++v) {
      const int pm  = mtile * 16 + khalf * 8 + v;
      const int pyo = pm >> 3, pxo = pm & 7;
      const int n   = (nt0 + j) * 16 + mrow;
      Y[(size_t)((((b * DOUT + zo) * HH) + (y0 + pyo)) * WW + (x0 + pxo)) * CO + n] = acc[j][v];
    }
  }
}

// ---------------------------------------------------------------------------
// Helpers: zero-fill, voxel scatter, mask propagation, masked BN, finalize.
// ---------------------------------------------------------------------------
__global__ void fillz(float* __restrict__ p, long n) {
  for (long i = (long)blockIdx.x * blockDim.x + threadIdx.x; i < n;
       i += (long)gridDim.x * blockDim.x) p[i] = 0.0f;
}

__global__ void scatter_voxels(const float* __restrict__ feats, const int* __restrict__ coors,
                               float* __restrict__ grid, float* __restrict__ mask, int n) {
  int i = blockIdx.x * blockDim.x + threadIdx.x;
  if (i >= n) return;
  const int b = coors[i * 4 + 0], z = coors[i * 4 + 1];
  const int y = coors[i * 4 + 2], x = coors[i * 4 + 3];
  const size_t pos = (((size_t)b * 21 + z) * HH + y) * WW + x;
  mask[pos] = 1.0f;
  const float4* s = (const float4*)&feats[i * 16];
  float4* d = (float4*)&grid[pos * 16];
  d[0] = s[0]; d[1] = s[1]; d[2] = s[2]; d[3] = s[3];
}

template<int SD, int PDZ, int KD, int KH, int KW>
__global__ void mask_upd(const float* __restrict__ Mi, float* __restrict__ Mo,
                         int DIN, int DOUT) {
  int i = blockIdx.x * blockDim.x + threadIdx.x;
  const int total = BB * DOUT * HH * WW;
  if (i >= total) return;
  int x = i % WW; int r = i / WW;
  int y = r % HH; r /= HH;
  int zo = r % DOUT; int b = r / DOUT;
  float m = 0.0f;
  for (int kd = 0; kd < KD; ++kd) {
    int iz = zo * SD - PDZ + kd;
    if (iz < 0 || iz >= DIN) continue;
    for (int kh = 0; kh < KH; ++kh) {
      int iy = y - ((KH > 1) ? 1 : 0) + kh;
      if (iy < 0 || iy >= HH) continue;
      for (int kw = 0; kw < KW; ++kw) {
        int ix = x - ((KW > 1) ? 1 : 0) + kw;
        if (ix < 0 || ix >= WW) continue;
        if (Mi[((size_t)(b * DIN + iz) * HH + iy) * WW + ix] > 0.0f) m = 1.0f;
      }
    }
  }
  Mo[i] = m;
}

// stats layout: st[0..63]=sum, st[64..127]=sumsq, st[128]=count
__global__ void bn_stats(const float* __restrict__ X, const float* __restrict__ Mk,
                         float* __restrict__ st, long npos, int CO) {
  __shared__ float ss[64], sq[64], sc;
  for (int k = threadIdx.x; k < 64; k += blockDim.x) { ss[k] = 0.0f; sq[k] = 0.0f; }
  if (threadIdx.x == 0) sc = 0.0f;
  __syncthreads();
  const int c = threadIdx.x % CO;
  const int g = threadIdx.x / CO;
  const int ng = blockDim.x / CO;
  float s = 0.0f, q = 0.0f, cn = 0.0f;
  for (long pos = (long)blockIdx.x * ng + g; pos < npos; pos += (long)gridDim.x * ng) {
    float m = Mk[pos];
    float v = X[pos * CO + c] * m;
    s += v; q += v * v;
    if (c == 0) cn += m;
  }
  atomicAdd(&ss[c], s);
  atomicAdd(&sq[c], q);
  if (c == 0) atomicAdd(&sc, cn);
  __syncthreads();
  for (int k = threadIdx.x; k < CO; k += blockDim.x) {
    atomicAdd(&st[k], ss[k]);
    atomicAdd(&st[64 + k], sq[k]);
  }
  if (threadIdx.x == 0) atomicAdd(&st[128], sc);
}

__global__ void bn_apply(float* __restrict__ X, const float* __restrict__ Mk,
                         const float* __restrict__ st, const float* __restrict__ gam,
                         const float* __restrict__ bet, long npos, int CO) {
  const int c = threadIdx.x % CO;
  const int g = threadIdx.x / CO;
  const int ng = blockDim.x / CO;
  const float cnt   = fmaxf(st[128], 1.0f);
  const float mean  = st[c] / cnt;
  const float var   = st[64 + c] / cnt - mean * mean;
  const float scale = rsqrtf(var + 1.0e-3f) * gam[c];
  const float shift = bet[c];
  for (long pos = (long)blockIdx.x * ng + g; pos < npos; pos += (long)gridDim.x * ng) {
    float m = Mk[pos];
    float v = X[pos * CO + c];
    X[pos * CO + c] = fmaxf((v - mean) * scale + shift, 0.0f) * m;
  }
}

// X [2,2,96,96,64] (NDHWC) -> out [2, 64*2, 96, 96]  (transpose 0,4,1,2,3 + reshape)
__global__ void finalize_out(const float* __restrict__ X, float* __restrict__ out) {
  int i = blockIdx.x * blockDim.x + threadIdx.x;
  const int total = BB * 2 * HH * WW * 64;
  if (i >= total) return;
  int c = i % 64; int r = i / 64;
  int x = r % WW; r /= WW;
  int y = r % HH; r /= HH;
  int z = r % 2;  int b = r / 2;
  out[(((size_t)(b * 128 + c * 2 + z) * HH + y) * WW + x)] = X[i];
}

// ---------------------------------------------------------------------------
// Host orchestration
// ---------------------------------------------------------------------------
extern "C" void kernel_launch(void* const* d_in, const int* in_sizes, int n_in,
                              void* d_out, int out_size, void* d_ws, size_t ws_size,
                              hipStream_t stream)
{
  (void)n_in; (void)out_size;
  const float* feats = (const float*)d_in[0];
  const int*   coors = (const int*)d_in[1];
  const int NVOX = in_sizes[0] / 16;

  const float* Wt[11]; const float* Gm[11]; const float* Bt[11];
  for (int i = 0; i < 11; ++i) {
    Wt[i] = (const float*)d_in[3 + 3 * i];
    Gm[i] = (const float*)d_in[4 + 3 * i];
    Bt[i] = (const float*)d_in[5 + 3 * i];
  }

  const size_t MAXE  = 12976128;                  // max layer tensor (2*11*96*96*64)
  const size_t NPOS0 = (size_t)BB * 21 * HH * WW; // 387072
  const size_t WBFE  = 27 * 64 * 64;              // max packed weight elems (bf16)
  float* ws = (float*)d_ws;
  float* buf[2] = { ws, ws + MAXE };
  float* msk[2] = { ws + 2 * MAXE, ws + 2 * MAXE + NPOS0 };
  float* st = ws + 2 * MAXE + 2 * NPOS0;
  ushort_t* wbf = (ushort_t*)(st + 192);
  if (ws_size < (2 * MAXE + 2 * NPOS0 + 192) * sizeof(float) + WBFE * sizeof(ushort_t)) return;

  // Dense grid + mask init, then scatter voxels.
  fillz<<<1024, 256, 0, stream>>>(buf[0], (long)NPOS0 * 16);
  fillz<<<128, 256, 0, stream>>>(msk[0], (long)NPOS0);
  scatter_voxels<<<(NVOX + 255) / 256, 256, 0, stream>>>(feats, coors, buf[0], msk[0], NVOX);

  int cur = 0, mcur = 0, nxt;
  long npos;

#define PREP(li_, CI_, CO_, NT_)                                                        \
  do {                                                                                  \
    const int cip_ = ((CI_) < 32) ? 32 : (CI_);                                         \
    const int tot_ = (NT_) * (CO_) * cip_;                                              \
    prep_weights<<<(tot_ + 255) / 256, 256, 0, stream>>>(Wt[li_], wbf, (CI_), (CO_),    \
                                                         cip_, (NT_));                  \
  } while (0)

#define BN_STEP(CO_, npos_, li_)                                                        \
  do {                                                                                  \
    fillz<<<1, 256, 0, stream>>>(st, 192);                                              \
    bn_stats<<<1024, 256, 0, stream>>>(buf[nxt], msk[mcur], st, (long)(npos_), (CO_));  \
    bn_apply<<<2048, 256, 0, stream>>>(buf[nxt], msk[mcur], st, Gm[li_], Bt[li_],       \
                                       (long)(npos_), (CO_));                           \
  } while (0)

  // L1: 16->32, stride 1, D=21
  nxt = cur ^ 1;
  PREP(0, 16, 32, 27);
  conv3d_wmma<16, 32, 1, 1, 3, 3, 3><<<BB * 21 * 144, 256, 0, stream>>>(buf[cur], wbf, buf[nxt], 21, 21);
  npos = (long)BB * 21 * HH * WW;
  BN_STEP(32, npos, 0); cur = nxt;

  // L2: 32->32
  nxt = cur ^ 1;
  PREP(1, 32, 32, 27);
  conv3d_wmma<32, 32, 1, 1, 3, 3, 3><<<BB * 21 * 144, 256, 0, stream>>>(buf[cur], wbf, buf[nxt], 21, 21);
  BN_STEP(32, npos, 1); cur = nxt;

  // L3: 32->64, stride 2 (pad_z=1), D 21->11
  nxt = cur ^ 1;
  PREP(2, 32, 64, 27);
  conv3d_wmma<32, 64, 2, 1, 3, 3, 3><<<BB * 11 * 144, 256, 0, stream>>>(buf[cur], wbf, buf[nxt], 21, 11);
  mask_upd<2, 1, 3, 3, 3><<<(BB * 11 * HH * WW + 255) / 256, 256, 0, stream>>>(msk[mcur], msk[mcur ^ 1], 21, 11);
  mcur ^= 1;
  npos = (long)BB * 11 * HH * WW;
  BN_STEP(64, npos, 2); cur = nxt;

  // L4-L6: 64->64, stride 1, D=11
  for (int li = 3; li <= 5; ++li) {
    nxt = cur ^ 1;
    PREP(li, 64, 64, 27);
    conv3d_wmma<64, 64, 1, 1, 3, 3, 3><<<BB * 11 * 144, 256, 0, stream>>>(buf[cur], wbf, buf[nxt], 11, 11);
    BN_STEP(64, npos, li); cur = nxt;
  }

  // L7: stride 2 (pad_z=0), D 11->5
  nxt = cur ^ 1;
  PREP(6, 64, 64, 27);
  conv3d_wmma<64, 64, 2, 0, 3, 3, 3><<<BB * 5 * 144, 256, 0, stream>>>(buf[cur], wbf, buf[nxt], 11, 5);
  mask_upd<2, 0, 3, 3, 3><<<(BB * 5 * HH * WW + 255) / 256, 256, 0, stream>>>(msk[mcur], msk[mcur ^ 1], 11, 5);
  mcur ^= 1;
  npos = (long)BB * 5 * HH * WW;
  BN_STEP(64, npos, 6); cur = nxt;

  // L8-L10: 64->64, stride 1, D=5
  for (int li = 7; li <= 9; ++li) {
    nxt = cur ^ 1;
    PREP(li, 64, 64, 27);
    conv3d_wmma<64, 64, 1, 1, 3, 3, 3><<<BB * 5 * 144, 256, 0, stream>>>(buf[cur], wbf, buf[nxt], 5, 5);
    BN_STEP(64, npos, li); cur = nxt;
  }

  // L11: (3,1,1) stride 2, pad 0, D 5->2
  nxt = cur ^ 1;
  PREP(10, 64, 64, 3);
  conv3d_wmma<64, 64, 2, 0, 3, 1, 1><<<BB * 2 * 144, 256, 0, stream>>>(buf[cur], wbf, buf[nxt], 5, 2);
  mask_upd<2, 0, 3, 1, 1><<<(BB * 2 * HH * WW + 255) / 256, 256, 0, stream>>>(msk[mcur], msk[mcur ^ 1], 5, 2);
  mcur ^= 1;
  npos = (long)BB * 2 * HH * WW;
  BN_STEP(64, npos, 10); cur = nxt;

  // Transpose (0,4,1,2,3) + reshape -> d_out
  finalize_out<<<(int)((npos * 64 + 255) / 256), 256, 0, stream>>>(buf[cur], (float*)d_out);

#undef PREP
#undef BN_STEP
}
